// model_44865228374620
// MI455X (gfx1250) — compile-verified, matrix operand order checked by reading, not run on hardware
//
#include <hip/hip_runtime.h>
#include <math.h>

typedef _Float16 half_t;
typedef __attribute__((ext_vector_type(16))) _Float16 v16h;
typedef __attribute__((ext_vector_type(8)))  _Float16 v8h;
typedef __attribute__((ext_vector_type(8)))  float    v8f;

#define BSZ   512
#define TCOND 1024
#define TTOT  1048   // TCOND + FUT(24)

// ---- helpers -------------------------------------------------------------

__device__ __forceinline__ v16h cat8(v8h lo, v8h hi) {
  v16h r;
#pragma unroll
  for (int i = 0; i < 8; ++i) { r[i] = lo[i]; r[i + 8] = hi[i]; }
  return r;
}

// A-fragment (16x32 f16) per CDNA5 layout: lane group g holds K runs
// [kc*32 + g*8 .. +7] and [kc*32 + 16 + g*8 .. +7] for row M = lane%16.
__device__ __forceinline__ v16h load_a(const _Float16* A, int ldk, int kc, int lane) {
  const _Float16* p = A + (lane & 15) * ldk + kc * 32 + ((lane >> 4) << 3);
  return cat8(*(const v8h*)p, *(const v8h*)(p + 16));
}

// CDNA5 V_TANH_F32 (TRANS op) when available.
__device__ __forceinline__ float fast_tanh(float x) {
#if __has_builtin(__builtin_amdgcn_tanhf)
  return __builtin_amdgcn_tanhf(x);
#else
  return fmaf(2.0f, __builtin_amdgcn_rcpf(1.0f + __expf(-2.0f * x)), -1.0f);
#endif
}
__device__ __forceinline__ float fast_sig(float x) {
  return fmaf(0.5f, fast_tanh(0.5f * x), 0.5f);
}

// ---- one-time weight packing ---------------------------------------------
// Gate rows PERMUTED so wave w owns all 4 gates of hidden units [4w,4w+4):
//   packed row p: w=p/16, c=p%16, original row = (c/4)*64 + 4w + (c%4)
// W0c[256][96]  : K 0..4 = W_ih0, 5..68 = W_hh0, 69..95 = 0
// W1c/W2c[256][128]: K 0..63 = W_ih, 64..127 = W_hh
// bias[3][256] stays in ORIGINAL layout. wh[16][64]: rows 0/1 = W_mean/W_std.
__global__ void pack_weights(const float* __restrict__ Wih0, const float* __restrict__ Whh0,
                             const float* __restrict__ bih0, const float* __restrict__ bhh0,
                             const float* __restrict__ Wih1, const float* __restrict__ Whh1,
                             const float* __restrict__ bih1, const float* __restrict__ bhh1,
                             const float* __restrict__ Wih2, const float* __restrict__ Whh2,
                             const float* __restrict__ bih2, const float* __restrict__ bhh2,
                             const float* __restrict__ Wm, const float* __restrict__ Ws,
                             half_t* __restrict__ w0c, half_t* __restrict__ w1c,
                             half_t* __restrict__ w2c, float* __restrict__ bias,
                             half_t* __restrict__ wh) {
  const int p = blockIdx.x;  // packed gate row 0..255
  const int c = p & 15, w = p >> 4;
  const int row = (c >> 2) * 64 + (w << 2) + (c & 3);  // original gate row
  for (int k = threadIdx.x; k < 96; k += blockDim.x) {
    float v = 0.0f;
    if (k < 5)       v = Wih0[row * 5 + k];
    else if (k < 69) v = Whh0[row * 64 + (k - 5)];
    w0c[p * 96 + k] = (half_t)v;
  }
  for (int k = threadIdx.x; k < 128; k += blockDim.x) {
    float v1 = (k < 64) ? Wih1[row * 64 + k] : Whh1[row * 64 + (k - 64)];
    float v2 = (k < 64) ? Wih2[row * 64 + k] : Whh2[row * 64 + (k - 64)];
    w1c[p * 128 + k] = (half_t)v1;
    w2c[p * 128 + k] = (half_t)v2;
  }
  if (threadIdx.x == 0) {
    bias[0 * 256 + p] = bih0[p] + bhh0[p];
    bias[1 * 256 + p] = bih1[p] + bhh1[p];
    bias[2 * 256 + p] = bih2[p] + bhh2[p];
  }
  if (p < 16) {
    for (int k = threadIdx.x; k < 64; k += blockDim.x) {
      float v = (p == 0) ? Wm[k] : (p == 1) ? Ws[k] : 0.0f;
      wh[p * 64 + k] = (half_t)v;
    }
  }
}

// ---- main sequential LSTM kernel -----------------------------------------
// grid = 32 (batch tiles of 16 rows), block = 544 threads = 17 wave32 waves.
// Waves 0..15: gate WMMAs + same-wave cell updates (units [4w,4w+4)).
// Wave 16: output head (2 WMMAs) + global I/O + x prefetch, overlapped.
// A-buffers are parity double-buffered: layer l at step t reads buffer t&1;
// cells write next-layer h into the current buffer (consumed after the next
// barrier) and recurrent h into the other buffer (consumed at t+1) -> no WAR.
// 3 barriers per conditioning step; +1 during the 24 autoregressive steps.
__global__ void __launch_bounds__(544, 1)
lstm_forecast(const float* __restrict__ input,   // [512][1024]
              const float* __restrict__ cov,     // [512][1048][4]
              const float* __restrict__ bm, const float* __restrict__ bs,
              const half_t* __restrict__ w0c, const half_t* __restrict__ w1c,
              const half_t* __restrict__ w2c, const float* __restrict__ bias,
              const half_t* __restrict__ wh,
              float* __restrict__ out)           // [512][1048][1][2]
{
  __shared__ __align__(16) _Float16 A0[2][16][96];   // [x(5) | h0(64) | pad]
  __shared__ __align__(16) _Float16 A1[2][16][128];  // [h0 | h1]
  __shared__ __align__(16) _Float16 A2[2][16][128];  // [h1 | h2]
  __shared__ float Gp[16][16][18];                   // per-wave gate tiles
  __shared__ float HD[16][2];                        // head (mean, std_raw)

  const int tid  = threadIdx.x;
  const int lane = tid & 31;
  const int wave = tid >> 5;             // 0..16
  const int b0   = blockIdx.x << 4;      // batch tile base

  for (int i = tid; i < 2 * 16 * 96;  i += 544) (&A0[0][0][0])[i] = (half_t)0.0f;
  for (int i = tid; i < 2 * 16 * 128; i += 544) (&A1[0][0][0])[i] = (half_t)0.0f;
  for (int i = tid; i < 2 * 16 * 128; i += 544) (&A2[0][0][0])[i] = (half_t)0.0f;

  // ---- weight fragments in registers (once) ----
  const int nrow = ((wave & 15) << 4) + (lane & 15);
  const int g8   = (lane >> 4) << 3;
  v16h wf0[3], wf1[4], wf2[4], whf[2];
  float bm0 = 0.0f, bs0 = 0.0f;
  if (wave < 16) {
#pragma unroll
    for (int kc = 0; kc < 3; ++kc) {
      const half_t* p = w0c + nrow * 96 + kc * 32 + g8;
      wf0[kc] = cat8(*(const v8h*)p, *(const v8h*)(p + 16));
    }
#pragma unroll
    for (int kc = 0; kc < 4; ++kc) {
      const half_t* p = w1c + nrow * 128 + kc * 32 + g8;
      const half_t* q = w2c + nrow * 128 + kc * 32 + g8;
      wf1[kc] = cat8(*(const v8h*)p, *(const v8h*)(p + 16));
      wf2[kc] = cat8(*(const v8h*)q, *(const v8h*)(q + 16));
    }
  } else {
#pragma unroll
    for (int kc = 0; kc < 2; ++kc) {   // head B matrix [W_mean;W_std;0..]
      const half_t* p = wh + (lane & 15) * 64 + kc * 32 + g8;
      whf[kc] = cat8(*(const v8h*)p, *(const v8h*)(p + 16));
    }
    bm0 = bm[0];
    bs0 = bs[0];
  }

  // ---- per-lane cell assignment (gate waves) ----
  const int jl = lane >> 3;               // unit within wave
  const int j  = ((wave & 15) << 2) + jl; // hidden unit 0..63
  const int m0 = (lane << 1) & 15;        // batch rows m0, m0+1
  const int mb = (lane >> 4) << 3;        // accumulator row base
  const int nc = lane & 15;               // packed gate col
  float cst[3][2] = {{0.f, 0.f}, {0.f, 0.f}, {0.f, 0.f}};
  float bgi[3], bgf[3], bgg[3], bgo[3];
  if (wave < 16) {
#pragma unroll
    for (int l = 0; l < 3; ++l) {
      bgi[l] = bias[l * 256 + j];
      bgf[l] = bias[l * 256 + 64 + j];
      bgg[l] = bias[l * 256 + 128 + j];
      bgo[l] = bias[l * 256 + 192 + j];
    }
  }

  // x(t=0) into buffer 0
  if (tid < 16) {
    const int b = b0 + tid;
    const float* cv = cov + (size_t)b * TTOT * 4;
    A0[0][tid][0] = (half_t)input[(size_t)b * TCOND];
    A0[0][tid][1] = (half_t)cv[0];
    A0[0][tid][2] = (half_t)cv[1];
    A0[0][tid][3] = (half_t)cv[2];
    A0[0][tid][4] = (half_t)cv[3];
  }

  // head for step u (wave 16 only). h2(u) lives in A2[(u+1)&1] cols 64..127.
  auto do_head = [&](int u) {
    _Float16 (*A2h)[96 + 32] = A2[(u + 1) & 1];
    v16h a0 = load_a(&A2h[0][0], 128, 2, lane);
    v16h a1 = load_a(&A2h[0][0], 128, 3, lane);
    v8f acc = {0.f, 0.f, 0.f, 0.f, 0.f, 0.f, 0.f, 0.f};
    acc = __builtin_amdgcn_wmma_f32_16x16x32_f16(false, a0, false, whf[0],
                                                 (short)0, acc, false, false);
    acc = __builtin_amdgcn_wmma_f32_16x16x32_f16(false, a1, false, whf[1],
                                                 (short)0, acc, false, false);
    if (nc < 2) {
#pragma unroll
      for (int v = 0; v < 8; ++v) HD[mb + v][nc] = acc[v];
    }
    if (lane < 16) {
      float m = HD[lane][0] + bm0;
      float s = HD[lane][1] + bs0;
      float sp = log1pf(expf(s));          // softplus with inf guard
      if (__builtin_isinf(sp)) sp = s;
      const size_t o = ((size_t)(b0 + lane) * TTOT + u) * 2;
      out[o]     = m;
      out[o + 1] = sp;
      const int tn = u + 1;
      if (tn >= TCOND && tn < TTOT) {      // autoregressive feedback x(tn)
        const float* cv = cov + ((size_t)(b0 + lane) * TTOT + tn) * 4;
        _Float16 (*A0x)[96] = A0[tn & 1];
        A0x[lane][0] = (half_t)m;
        A0x[lane][1] = (half_t)cv[0];
        A0x[lane][2] = (half_t)cv[1];
        A0x[lane][3] = (half_t)cv[2];
        A0x[lane][4] = (half_t)cv[3];
      }
    }
  };

  __syncthreads();

#pragma unroll 1
  for (int t = 0; t < TTOT; ++t) {
    _Float16 (*A0r)[96]  = A0[t & 1];
    _Float16 (*A0w)[96]  = A0[(t + 1) & 1];
    _Float16 (*A1r)[128] = A1[t & 1];
    _Float16 (*A1w)[128] = A1[(t + 1) & 1];
    _Float16 (*A2r)[128] = A2[t & 1];
    _Float16 (*A2w)[128] = A2[(t + 1) & 1];

    if (wave == 16 && t > 0) do_head(t - 1);   // overlaps with gate waves

    if (t >= TCOND) __syncthreads();           // release feedback x(t)

    if (wave < 16) {
      // ========== layer 0: gates = [x|h0] (16x96) x W0^T ==========
      v16h a0 = load_a(&A0r[0][0], 96, 0, lane);
      v16h a1 = load_a(&A0r[0][0], 96, 1, lane);
      v16h a2 = load_a(&A0r[0][0], 96, 2, lane);
      v8f acc = {0.f, 0.f, 0.f, 0.f, 0.f, 0.f, 0.f, 0.f};
      acc = __builtin_amdgcn_wmma_f32_16x16x32_f16(false, a0, false, wf0[0],
                                                   (short)0, acc, false, false);
      acc = __builtin_amdgcn_wmma_f32_16x16x32_f16(false, a1, false, wf0[1],
                                                   (short)0, acc, false, false);
      acc = __builtin_amdgcn_wmma_f32_16x16x32_f16(false, a2, false, wf0[2],
                                                   (short)0, acc, false, false);
#pragma unroll
      for (int v = 0; v < 8; ++v) Gp[wave][mb + v][nc] = acc[v];
#pragma unroll
      for (int rr = 0; rr < 2; ++rr) {  // same-wave cell 0
        const int m = m0 + rr;
        float gi = Gp[wave][m][jl]      + bgi[0];
        float gf = Gp[wave][m][4 + jl]  + bgf[0];
        float gg = Gp[wave][m][8 + jl]  + bgg[0];
        float go = Gp[wave][m][12 + jl] + bgo[0];
        float c  = fast_sig(gf) * cst[0][rr] + fast_sig(gi) * fast_tanh(gg);
        float h  = fast_sig(go) * fast_tanh(c);
        cst[0][rr] = c;
        half_t hf = (half_t)h;
        A0w[m][5 + j] = hf;   // recurrent input for t+1 (other buffer)
        A1r[m][j]     = hf;   // layer-1 input for this step (after barrier)
      }
    } else if (t + 1 < TCOND && lane < 16) {   // prefetch x(t+1) from global
      const int b = b0 + lane;
      const float* cv = cov + ((size_t)b * TTOT + (t + 1)) * 4;
      A0w[lane][0] = (half_t)input[(size_t)b * TCOND + (t + 1)];
      A0w[lane][1] = (half_t)cv[0];
      A0w[lane][2] = (half_t)cv[1];
      A0w[lane][3] = (half_t)cv[2];
      A0w[lane][4] = (half_t)cv[3];
    }
    __syncthreads();  // B1

    if (wave < 16) {
      // ========== layer 1: gates = [h0|h1] (16x128) x W1^T ==========
      v16h a0 = load_a(&A1r[0][0], 128, 0, lane);
      v16h a1 = load_a(&A1r[0][0], 128, 1, lane);
      v16h a2 = load_a(&A1r[0][0], 128, 2, lane);
      v16h a3 = load_a(&A1r[0][0], 128, 3, lane);
      v8f acc = {0.f, 0.f, 0.f, 0.f, 0.f, 0.f, 0.f, 0.f};
      acc = __builtin_amdgcn_wmma_f32_16x16x32_f16(false, a0, false, wf1[0],
                                                   (short)0, acc, false, false);
      acc = __builtin_amdgcn_wmma_f32_16x16x32_f16(false, a1, false, wf1[1],
                                                   (short)0, acc, false, false);
      acc = __builtin_amdgcn_wmma_f32_16x16x32_f16(false, a2, false, wf1[2],
                                                   (short)0, acc, false, false);
      acc = __builtin_amdgcn_wmma_f32_16x16x32_f16(false, a3, false, wf1[3],
                                                   (short)0, acc, false, false);
#pragma unroll
      for (int v = 0; v < 8; ++v) Gp[wave][mb + v][nc] = acc[v];
#pragma unroll
      for (int rr = 0; rr < 2; ++rr) {  // cell 1
        const int m = m0 + rr;
        float gi = Gp[wave][m][jl]      + bgi[1];
        float gf = Gp[wave][m][4 + jl]  + bgf[1];
        float gg = Gp[wave][m][8 + jl]  + bgg[1];
        float go = Gp[wave][m][12 + jl] + bgo[1];
        float c  = fast_sig(gf) * cst[1][rr] + fast_sig(gi) * fast_tanh(gg);
        float h  = fast_sig(go) * fast_tanh(c);
        cst[1][rr] = c;
        half_t hf = (half_t)h;
        A1w[m][64 + j] = hf;  // recurrent for t+1
        A2r[m][j]      = hf;  // layer-2 input for this step
      }
    }
    __syncthreads();  // B2

    if (wave < 16) {
      // ========== layer 2: gates = [h1|h2] (16x128) x W2^T ==========
      v16h a0 = load_a(&A2r[0][0], 128, 0, lane);
      v16h a1 = load_a(&A2r[0][0], 128, 1, lane);
      v16h a2 = load_a(&A2r[0][0], 128, 2, lane);
      v16h a3 = load_a(&A2r[0][0], 128, 3, lane);
      v8f acc = {0.f, 0.f, 0.f, 0.f, 0.f, 0.f, 0.f, 0.f};
      acc = __builtin_amdgcn_wmma_f32_16x16x32_f16(false, a0, false, wf2[0],
                                                   (short)0, acc, false, false);
      acc = __builtin_amdgcn_wmma_f32_16x16x32_f16(false, a1, false, wf2[1],
                                                   (short)0, acc, false, false);
      acc = __builtin_amdgcn_wmma_f32_16x16x32_f16(false, a2, false, wf2[2],
                                                   (short)0, acc, false, false);
      acc = __builtin_amdgcn_wmma_f32_16x16x32_f16(false, a3, false, wf2[3],
                                                   (short)0, acc, false, false);
#pragma unroll
      for (int v = 0; v < 8; ++v) Gp[wave][mb + v][nc] = acc[v];
#pragma unroll
      for (int rr = 0; rr < 2; ++rr) {  // cell 2
        const int m = m0 + rr;
        float gi = Gp[wave][m][jl]      + bgi[2];
        float gf = Gp[wave][m][4 + jl]  + bgf[2];
        float gg = Gp[wave][m][8 + jl]  + bgg[2];
        float go = Gp[wave][m][12 + jl] + bgo[2];
        float c  = fast_sig(gf) * cst[2][rr] + fast_sig(gi) * fast_tanh(gg);
        float h  = fast_sig(go) * fast_tanh(c);
        cst[2][rr] = c;
        A2w[m][64 + j] = (half_t)h;  // h2(t): head + recurrent for t+1
      }
    }
    __syncthreads();  // B3
  }

  if (wave == 16) do_head(TTOT - 1);   // final output
}

// ---- launcher -------------------------------------------------------------
extern "C" void kernel_launch(void* const* d_in, const int* in_sizes, int n_in,
                              void* d_out, int out_size, void* d_ws, size_t ws_size,
                              hipStream_t stream) {
  (void)in_sizes; (void)n_in; (void)out_size; (void)ws_size;

  const float* input = (const float*)d_in[0];
  const float* cov   = (const float*)d_in[1];
  const float* Wm    = (const float*)d_in[2];
  const float* bm    = (const float*)d_in[3];
  const float* Ws    = (const float*)d_in[4];
  const float* bs    = (const float*)d_in[5];
  // d_in[6] = future (compile-time constant 24 here)
  const float* Wih0 = (const float*)d_in[7];
  const float* Whh0 = (const float*)d_in[8];
  const float* bih0 = (const float*)d_in[9];
  const float* bhh0 = (const float*)d_in[10];
  const float* Wih1 = (const float*)d_in[11];
  const float* Whh1 = (const float*)d_in[12];
  const float* bih1 = (const float*)d_in[13];
  const float* bhh1 = (const float*)d_in[14];
  const float* Wih2 = (const float*)d_in[15];
  const float* Whh2 = (const float*)d_in[16];
  const float* bih2 = (const float*)d_in[17];
  const float* bhh2 = (const float*)d_in[18];

  half_t* w0c = (half_t*)d_ws;               // 256*96 halves
  half_t* w1c = w0c + 256 * 96;              // 256*128
  half_t* w2c = w1c + 256 * 128;             // 256*128
  float*  bias = (float*)(w2c + 256 * 128);  // 3*256 f32
  half_t* whh  = (half_t*)(bias + 3 * 256);  // 16*64 halves (head)

  pack_weights<<<dim3(256), dim3(128), 0, stream>>>(
      Wih0, Whh0, bih0, bhh0, Wih1, Whh1, bih1, bhh1,
      Wih2, Whh2, bih2, bhh2, Wm, Ws, w0c, w1c, w2c, bias, whh);

  lstm_forecast<<<dim3(32), dim3(544), 0, stream>>>(
      input, cov, bm, bs, w0c, w1c, w2c, bias, whh, (float*)d_out);
}
